// Fastformer_44727789420894
// MI455X (gfx1250) — matrix operand order repaired and checked by reading, not compiled, not run on hardware
//
#include <hip/hip_runtime.h>
#include <hip/hip_bf16.h>
#include <math.h>

// ---------------------------------------------------------------- constants
#define BB   8
#define NN   4096
#define FF   512          // DIM (K of first GEMMs)
#define DD   512          // decode dim
#define NCH  16           // N split chunks for softmax partials
#define NPC  (NN / NCH)   // rows per chunk = 256
static const float SCALE_F = 0.044194173824159216f;   // 512^-0.5

typedef unsigned short ushort_t;
typedef __attribute__((ext_vector_type(16))) __bf16    v16bf;
typedef __attribute__((ext_vector_type(8)))  float     v8f;
typedef __attribute__((ext_vector_type(8)))  unsigned short ushort8;
typedef __attribute__((ext_vector_type(4)))  unsigned short ushort4v;

union bfcast { ushort8 h[2]; v16bf v; };

__device__ __forceinline__ v8f zero8() {
    v8f z;
    #pragma unroll
    for (int i = 0; i < 8; ++i) z[i] = 0.0f;
    return z;
}

__device__ __forceinline__ unsigned short f2bf(float f) {
    unsigned u = __float_as_uint(f);
    u += 0x7FFFu + ((u >> 16) & 1u);        // round-to-nearest-even
    return (unsigned short)(u >> 16);
}

// A fragment (16x32 bf16, row-major src): lane r=lane&15 -> row, kh=lane>>4.
// elements 0..7 at K = kh*8 + (0..7); elements 8..15 at K = 16 + kh*8 + (0..7)
__device__ __forceinline__ v16bf load_frag_a(const ushort_t* p) {
    bfcast u;
    u.h[0] = *(const ushort8*)(p);
    u.h[1] = *(const ushort8*)(p + 16);
    return u.v;
}
// B fragment (32x16 bf16): lane c=lane&15 -> col (a row of W), kh=lane>>4.
// elements 0..7 at K = kh*8 + (0..7); elements 8..15 at K = 16 + kh*8 + (0..7)
__device__ __forceinline__ v16bf load_frag_b(const ushort_t* p) {
    bfcast u;
    u.h[0] = *(const ushort8*)(p);
    u.h[1] = *(const ushort8*)(p + 16);
    return u.v;
}

__device__ __forceinline__ v8f wmma_bf16(v16bf a, v16bf b, v8f c) {
    return __builtin_amdgcn_wmma_f32_16x16x32_bf16(
        false, a, false, b, (short)0, c, false, false);
}

// ---------------------------------------------------------------- kernel 1: f32 -> bf16
__global__ __launch_bounds__(256) void k_f32_to_bf16(const float* __restrict__ in,
                                                     ushort_t* __restrict__ out,
                                                     unsigned long long n4) {
    unsigned long long i = ((unsigned long long)blockIdx.x * blockDim.x + threadIdx.x);
    if (i >= n4) return;
    i <<= 2;
    float4 f = *(const float4*)(in + i);
    ushort4v o;
    o.x = f2bf(f.x); o.y = f2bf(f.y); o.z = f2bf(f.z); o.w = f2bf(f.w);
    *(ushort4v*)(out + i) = o;
}

// ---------------------------------------------------------------- kernel 2: fused QKV GEMM
// grid (NN/128, BB, DD/64), block 256 (8 waves).
// All 8 waves share one 64-col strip (identical B-fragment address stream ->
// WGP$/L0 temporal hits); wave w takes rows (blockIdx.x*8+w)*16 .. +15.
__global__ __launch_bounds__(256) void k_qkv_wmma(
    const ushort_t* __restrict__ xb,
    const ushort_t* __restrict__ wq, const ushort_t* __restrict__ wk,
    const ushort_t* __restrict__ wv,
    float* __restrict__ q, float* __restrict__ k, float* __restrict__ v) {
    const int lane = threadIdx.x & 31;
    const int wave = threadIdx.x >> 5;
    const int b    = blockIdx.y;
    const int row0 = (blockIdx.x * 8 + wave) << 4;
    const int col0 = blockIdx.z << 6;
    const int r    = lane & 15;
    const int kh   = lane >> 4;

    const ushort_t* ap = xb + ((size_t)b * NN + row0 + r) * FF + kh * 8;

    v8f aq[4], ak[4], av[4];
    #pragma unroll
    for (int t = 0; t < 4; ++t) { aq[t] = zero8(); ak[t] = zero8(); av[t] = zero8(); }

    for (int kk = 0; kk < FF; kk += 32) {
        if (kk + 32 < FF) __builtin_prefetch(ap + kk + 32, 0, 0);  // global_prefetch_b8
        v16bf a = load_frag_a(ap + kk);
        #pragma unroll
        for (int t = 0; t < 4; ++t) {
            const size_t wrow = (size_t)(col0 + t * 16 + r) * FF + kk + kh * 8;
            aq[t] = wmma_bf16(a, load_frag_b(wq + wrow), aq[t]);
            ak[t] = wmma_bf16(a, load_frag_b(wk + wrow), ak[t]);
            av[t] = wmma_bf16(a, load_frag_b(wv + wrow), av[t]);
        }
    }

    const size_t ob = (size_t)b * NN + row0 + kh * 8;   // D-matrix: M = e + 8*kh
    #pragma unroll
    for (int t = 0; t < 4; ++t) {
        #pragma unroll
        for (int e = 0; e < 8; ++e) {
            const size_t idx = (ob + e) * DD + col0 + t * 16 + r;
            q[idx] = aq[t][e];
            k[idx] = ak[t][e];
            v[idx] = av[t][e];
        }
    }
}

// ---------------------------------------------------------------- kernel 3: alpha partials
// grid (DD/256, BB, NCH). Online softmax over the chunk: keep (max m, sum s, weighted t).
__global__ __launch_bounds__(256) void k_alpha_partial(
    const float* __restrict__ q, const float* __restrict__ w_alpha,
    float* __restrict__ part) {
    const int d  = blockIdx.x * 256 + threadIdx.x;
    const int b  = blockIdx.y;
    const int nc = blockIdx.z;
    const float wa = w_alpha[d] * SCALE_F;
    const float* qp = q + ((size_t)b * NN + (size_t)nc * NPC) * DD + d;
    float m = -INFINITY, s = 0.f, t = 0.f;
    for (int n = 0; n < NPC; ++n) {
        const float qv = qp[(size_t)n * DD];
        const float l  = qv * wa;
        const float mn = fmaxf(m, l);
        const float c  = __expf(m - mn);
        const float e  = __expf(l - mn);
        s = s * c + e;
        t = t * c + qv * e;
        m = mn;
    }
    float* o = part + (((size_t)b * NCH + nc) * DD + d) * 3;
    o[0] = m; o[1] = s; o[2] = t;
}

// ---------------------------------------------------------------- kernel 4: beta partials
__global__ __launch_bounds__(256) void k_beta_partial(
    const float* __restrict__ kmat, const float* __restrict__ gq,
    const float* __restrict__ w_beta, float* __restrict__ part) {
    const int d  = blockIdx.x * 256 + threadIdx.x;
    const int b  = blockIdx.y;
    const int nc = blockIdx.z;
    const float g  = gq[b * DD + d];
    const float wb = w_beta[d] * SCALE_F;
    const float* kp = kmat + ((size_t)b * NN + (size_t)nc * NPC) * DD + d;
    float m = -INFINITY, s = 0.f, t = 0.f;
    for (int n = 0; n < NPC; ++n) {
        const float pv = g * kp[(size_t)n * DD];
        const float l  = pv * wb;
        const float mn = fmaxf(m, l);
        const float c  = __expf(m - mn);
        const float e  = __expf(l - mn);
        s = s * c + e;
        t = t * c + pv * e;
        m = mn;
    }
    float* o = part + (((size_t)b * NCH + nc) * DD + d) * 3;
    o[0] = m; o[1] = s; o[2] = t;
}

// ---------------------------------------------------------------- kernel: combine partials
// grid (BB*DD/256). g[b,d] = (sum q*softmax) = t_total / s_total.
__global__ __launch_bounds__(256) void k_softmax_combine(
    const float* __restrict__ part, float* __restrict__ g) {
    const int idx = blockIdx.x * 256 + threadIdx.x;   // b*DD + d
    const int b = idx >> 9;
    const int d = idx & (DD - 1);
    float m = -INFINITY, s = 0.f, t = 0.f;
    for (int nc = 0; nc < NCH; ++nc) {
        const float* p = part + (((size_t)b * NCH + nc) * DD + d) * 3;
        const float mi = p[0], si = p[1], ti = p[2];
        const float mn = fmaxf(m, mi);
        const float c0 = __expf(m - mn);
        const float c1 = __expf(mi - mn);
        s = s * c0 + si * c1;
        t = t * c0 + ti * c1;
        m = mn;
    }
    g[idx] = t / s;
}

// ---------------------------------------------------------------- kernel 5: kv = gk * v -> bf16
__global__ __launch_bounds__(256) void k_make_kvb(
    const float* __restrict__ v, const float* __restrict__ gk,
    ushort_t* __restrict__ kvb) {
    const size_t i = (((size_t)blockIdx.x * blockDim.x) + threadIdx.x) << 2;
    const size_t b = i >> 21;               // NN*DD = 2^21
    const size_t d = i & (DD - 1);
    float4 vv = *(const float4*)(v + i);
    const float* g = gk + b * DD + d;
    ushort4v o;
    o.x = f2bf(g[0] * vv.x);
    o.y = f2bf(g[1] * vv.y);
    o.z = f2bf(g[2] * vv.z);
    o.w = f2bf(g[3] * vv.w);
    *(ushort4v*)(kvb + i) = o;
}

// ---------------------------------------------------------------- kernel 6: out = kv @ Wr^T + q
// Same block->tile mapping as k_qkv_wmma. q already lives in qout (d_out);
// per-element read-modify-write by the owning thread, no cross-thread hazard.
__global__ __launch_bounds__(256) void k_out_wmma(
    const ushort_t* __restrict__ kvb, const ushort_t* __restrict__ wr,
    float* qout) {
    const int lane = threadIdx.x & 31;
    const int wave = threadIdx.x >> 5;
    const int b    = blockIdx.y;
    const int row0 = (blockIdx.x * 8 + wave) << 4;
    const int col0 = blockIdx.z << 6;
    const int r    = lane & 15;
    const int kh   = lane >> 4;

    const ushort_t* ap = kvb + ((size_t)b * NN + row0 + r) * DD + kh * 8;

    v8f acc[4];
    #pragma unroll
    for (int t = 0; t < 4; ++t) acc[t] = zero8();

    for (int kk = 0; kk < DD; kk += 32) {
        if (kk + 32 < DD) __builtin_prefetch(ap + kk + 32, 0, 0);
        v16bf a = load_frag_a(ap + kk);
        #pragma unroll
        for (int t = 0; t < 4; ++t) {
            const size_t wrow = (size_t)(col0 + t * 16 + r) * DD + kk + kh * 8;
            acc[t] = wmma_bf16(a, load_frag_b(wr + wrow), acc[t]);
        }
    }

    const size_t ob = (size_t)b * NN + row0 + kh * 8;
    #pragma unroll
    for (int t = 0; t < 4; ++t) {
        #pragma unroll
        for (int e = 0; e < 8; ++e) {
            const size_t idx = (ob + e) * DD + col0 + t * 16 + r;
            qout[idx] = qout[idx] + acc[t][e];
        }
    }
}

// ---------------------------------------------------------------- launch
extern "C" void kernel_launch(void* const* d_in, const int* in_sizes, int n_in,
                              void* d_out, int out_size, void* d_ws, size_t ws_size,
                              hipStream_t stream) {
    const float* x       = (const float*)d_in[0];
    const float* Wq      = (const float*)d_in[1];
    const float* Wk      = (const float*)d_in[2];
    const float* Wv      = (const float*)d_in[3];
    const float* Wr      = (const float*)d_in[4];
    const float* w_alpha = (const float*)d_in[5];
    const float* w_beta  = (const float*)d_in[6];
    (void)in_sizes; (void)n_in; (void)ws_size; (void)out_size;

    char* ws = (char*)d_ws;
    // workspace layout (bytes)
    const size_t XB_OFF   = 0;                                   // 32 MB  bf16 x
    const size_t WQB_OFF  = XB_OFF  + (size_t)BB * NN * FF * 2;  // 512 KB each
    const size_t WKB_OFF  = WQB_OFF + (size_t)DD * FF * 2;
    const size_t WVB_OFF  = WKB_OFF + (size_t)DD * FF * 2;
    const size_t WRB_OFF  = WVB_OFF + (size_t)DD * FF * 2;
    const size_t K_OFF    = WRB_OFF + (size_t)DD * DD * 2;       // 64 MB f32 k
    const size_t V_OFF    = K_OFF   + (size_t)BB * NN * DD * 4;  // 64 MB f32 v
    const size_t PART_OFF = V_OFF   + (size_t)BB * NN * DD * 4;  // 768 KB partials
    const size_t GQ_OFF   = PART_OFF+ (size_t)BB * NCH * DD * 3 * 4;
    const size_t GK_OFF   = GQ_OFF  + (size_t)BB * DD * 4;
    const size_t KVB_OFF  = GK_OFF  + (size_t)BB * DD * 4;       // 32 MB bf16 kv

    ushort_t* xb  = (ushort_t*)(ws + XB_OFF);
    ushort_t* wqb = (ushort_t*)(ws + WQB_OFF);
    ushort_t* wkb = (ushort_t*)(ws + WKB_OFF);
    ushort_t* wvb = (ushort_t*)(ws + WVB_OFF);
    ushort_t* wrb = (ushort_t*)(ws + WRB_OFF);
    float*    kf  = (float*)(ws + K_OFF);
    float*    vf  = (float*)(ws + V_OFF);
    float*    prt = (float*)(ws + PART_OFF);
    float*    gq  = (float*)(ws + GQ_OFF);
    float*    gk  = (float*)(ws + GK_OFF);
    ushort_t* kvb = (ushort_t*)(ws + KVB_OFF);
    float*    qo  = (float*)d_out;   // q lives here; final kernel does += on it

    // 1) f32 -> bf16 conversions
    {
        const unsigned long long nx4 = (unsigned long long)BB * NN * FF / 4;   // 4,194,304
        k_f32_to_bf16<<<dim3((unsigned)(nx4 / 256)), dim3(256), 0, stream>>>(x, xb, nx4);
        const unsigned long long nw4 = (unsigned long long)DD * FF / 4;        // 65,536
        k_f32_to_bf16<<<dim3((unsigned)(nw4 / 256)), dim3(256), 0, stream>>>(Wq, wqb, nw4);
        k_f32_to_bf16<<<dim3((unsigned)(nw4 / 256)), dim3(256), 0, stream>>>(Wk, wkb, nw4);
        k_f32_to_bf16<<<dim3((unsigned)(nw4 / 256)), dim3(256), 0, stream>>>(Wv, wvb, nw4);
        k_f32_to_bf16<<<dim3((unsigned)(nw4 / 256)), dim3(256), 0, stream>>>(Wr, wrb, nw4);
    }

    // 2) fused QKV GEMM (q -> d_out)
    k_qkv_wmma<<<dim3(NN / 128, BB, DD / 64), dim3(256), 0, stream>>>(xb, wqb, wkb, wvb, qo, kf, vf);

    // 3) alpha softmax reduction -> global_query
    k_alpha_partial<<<dim3(DD / 256, BB, NCH), dim3(256), 0, stream>>>(qo, w_alpha, prt);
    k_softmax_combine<<<dim3(BB * DD / 256), dim3(256), 0, stream>>>(prt, gq);

    // 4) beta softmax reduction on p = gq*k -> global_key
    k_beta_partial<<<dim3(DD / 256, BB, NCH), dim3(256), 0, stream>>>(kf, gq, w_beta, prt);
    k_softmax_combine<<<dim3(BB * DD / 256), dim3(256), 0, stream>>>(prt, gk);

    // 5) kv = gk * v -> bf16
    {
        const unsigned nkv = (unsigned)((size_t)BB * NN * DD / 4 / 256);       // 16384 blocks
        k_make_kvb<<<dim3(nkv), dim3(256), 0, stream>>>(vf, gk, kvb);
    }

    // 6) out = kv @ Wr^T + q  (in place on d_out)
    k_out_wmma<<<dim3(NN / 128, BB, DD / 64), dim3(256), 0, stream>>>(kvb, wrb, qo);
}
